// PositionGRUEmbedding_18545668784522
// MI455X (gfx1250) — matrix operand order repaired and checked by reading, not compiled, not run on hardware
//
#include <hip/hip_runtime.h>

// ---------------------------------------------------------------------------
// PositionGRUEmbedding for MI455X (gfx1250, wave32, WMMA).
// All GEMMs run through v_wmma_f32_16x16x32_bf16 (bf16 in, f32 accumulate).
// B=128, S=128, D=1024, DS=64.
// ---------------------------------------------------------------------------

typedef __attribute__((ext_vector_type(16))) __bf16 v16bf;
typedef __attribute__((ext_vector_type(8)))  __bf16 v8bf;
typedef __attribute__((ext_vector_type(8)))  float  v8f;

#define B_  128
#define S_  128
#define D_  1024
#define DS_ 64

// ---------------------------------------------------------------------------
// fp32 -> bf16 converter (grid-stride)
// ---------------------------------------------------------------------------
__global__ __launch_bounds__(256) void f32_to_bf16_kernel(
    const float* __restrict__ in, __bf16* __restrict__ out, size_t n) {
    size_t i = (size_t)blockIdx.x * blockDim.x + threadIdx.x;
    size_t stride = (size_t)gridDim.x * blockDim.x;
    for (; i < n; i += stride) out[i] = (__bf16)in[i];
}

// ---------------------------------------------------------------------------
// Generic WMMA GEMM:  C[M,N] = act( A[M,K] * Bw[N,K]^T + bias [+ C] )
//  - A, Bw are bf16, K-contiguous rows (lda/ldb in elements).
//  - Optional gather on A rows: row(m) = m*Sdim + fa[m*Sdim + step]  (GRU).
//  - Cf (f32) and/or Cb (bf16) outputs; accumulate adds existing Cf.
//  - act: 0 = none, 1 = exact GELU.
// Block = 256 threads = 8 waves arranged 4(M) x 2(N); wave tile = 32x64
// (2x4 tiles of v_wmma_f32_16x16x32_bf16 -> 8 WMMA per K=32 step against
//  6 x 32B fragment loads). Block tile = 128 x 128.
// Requires M % 32 == 0, N % 64 == 0, K % 32 == 0 (true for all calls here).
// ---------------------------------------------------------------------------
__global__ __launch_bounds__(256) void wmma_gemm_kernel(
    const __bf16* __restrict__ A, int lda,
    const __bf16* __restrict__ Bw, int ldb,
    const float* __restrict__ bias,
    float* __restrict__ Cf, __bf16* __restrict__ Cb, int ldc,
    int M, int N, int K,
    int accumulate, int act,
    const int* __restrict__ fa, int step, int Sdim) {

    const int wid  = threadIdx.x >> 5;
    const int lane = threadIdx.x & 31;
    const int wm = wid & 3;        // 4 waves along M
    const int wn = wid >> 2;       // 2 waves along N
    const int m0 = blockIdx.x * 128 + wm * 32;
    const int n0 = blockIdx.y * 128 + wn * 64;
    if (m0 >= M || n0 >= N) return;   // whole-wave uniform exit (EXEC stays full)

    const int half = lane >> 4;    // 0: lanes 0-15, 1: lanes 16-31
    const int l16  = lane & 15;

    // A-row pointers for the two 16-row tiles this wave owns.
    // A order used here (contiguous k, as for B): every lane holds row (lane&15); lanes 0-15 carry
    // K = [0..15], lanes 16-31 carry K = [16..31].
    const __bf16* arow[2];
#pragma unroll
    for (int i = 0; i < 2; ++i) {
        const int m = m0 + i * 16 + l16;
        size_t row;
        if (fa) row = (size_t)m * Sdim + (size_t)fa[(size_t)m * Sdim + step];
        else    row = (size_t)m;
        arow[i] = A + row * (size_t)lda;
    }
    // B (K x N) column pointers: column n = Bw row n (K-contiguous).
    // Lanes 0-15 carry K=[0..15], lanes 16-31 carry K=[16..31] of each tile.
    const __bf16* brow[4];
#pragma unroll
    for (int j = 0; j < 4; ++j)
        brow[j] = Bw + (size_t)(n0 + j * 16 + l16) * (size_t)ldb;

    v8f acc[2][4];
#pragma unroll
    for (int i = 0; i < 2; ++i)
#pragma unroll
        for (int j = 0; j < 4; ++j)
            acc[i][j] = (v8f){0.f, 0.f, 0.f, 0.f, 0.f, 0.f, 0.f, 0.f};

    for (int k0 = 0; k0 < K; k0 += 32) {
        v16bf afrag[2], bfrag[4];
#pragma unroll
        for (int i = 0; i < 2; ++i) {
            afrag[i] = *(const v16bf*)(arow[i] + k0 + half * 16);   // K 0-15 / 16-31
        }
#pragma unroll
        for (int j = 0; j < 4; ++j)
            bfrag[j] = *(const v16bf*)(brow[j] + k0 + half * 16);    // 32B load

        // Prefetch the streaming A operand two K-steps ahead (weights sit in
        // L2 already; A is the 32 MB streamed activation matrix).
        if (k0 + 64 < K) {
#pragma unroll
            for (int i = 0; i < 2; ++i)
                __builtin_prefetch(arow[i] + k0 + 64, 0, 0);
        }

#pragma unroll
        for (int i = 0; i < 2; ++i)
#pragma unroll
            for (int j = 0; j < 4; ++j)
                acc[i][j] = __builtin_amdgcn_wmma_f32_16x16x32_bf16(
                    false, afrag[i], false, bfrag[j],
                    (short)0, acc[i][j], false, false);
    }

    // Epilogue. f32 C/D layout: VGPR r of lanes 0-15 -> M=r, lanes 16-31 -> M=8+r,
    // N = lane&15 within each 16-wide tile.
#pragma unroll
    for (int i = 0; i < 2; ++i) {
#pragma unroll
        for (int j = 0; j < 4; ++j) {
            const int col = n0 + j * 16 + l16;
            const float bv = bias ? bias[col] : 0.0f;
#pragma unroll
            for (int r = 0; r < 8; ++r) {
                const int row = m0 + i * 16 + half * 8 + r;
                const size_t idx = (size_t)row * (size_t)ldc + col;
                float c = acc[i][j][r] + bv;
                if (accumulate) c += Cf[idx];
                if (act == 1)   c = 0.5f * c * (1.0f + erff(c * 0.7071067811865475f));
                if (Cf) Cf[idx] = c;
                if (Cb) Cb[idx] = (__bf16)c;
            }
        }
    }
}

// ---------------------------------------------------------------------------
// GRU gate kernel: one block per batch row b, 256 threads, 4 values/thread.
//   xr,xz,xn = gx_all[b, step] splits; hr,hz,hn = gh[b] splits (bh included).
//   r = sigmoid(LN0(xr+hr)); z = sigmoid(LN1(xz+hz));
//   n = tanh(LN2(xn + r*hn)); h = (1-z)*n + z*hx.
// ---------------------------------------------------------------------------
__global__ __launch_bounds__(256) void gru_gate_kernel(
    const float* __restrict__ gx_all,   // (B,S,3D), bx included
    const float* __restrict__ gh,       // (B,3D), bh included
    const float* __restrict__ ln_g,     // (3,D)
    const float* __restrict__ ln_b,     // (3,D)
    const int*   __restrict__ fa,       // (B,S)
    int step,
    float*  __restrict__ gru_f32,       // (B,S,D)
    __bf16* __restrict__ gru_bf) {      // (B,S,D)

    constexpr int TPB = 256, VPT = D_ / TPB;
    __shared__ float red[TPB];
    const int b = blockIdx.x, t = threadIdx.x;
    const int parent = fa[b * S_ + step];

    const float* gx  = gx_all + ((size_t)b * S_ + step) * (size_t)(3 * D_);
    const float* ghb = gh + (size_t)b * (3 * D_);
    const float* hxp = gru_f32 + ((size_t)b * S_ + parent) * (size_t)D_;

    float hx[VPT], r[VPT], z[VPT], n[VPT], pre[VPT];
#pragma unroll
    for (int k = 0; k < VPT; ++k) hx[k] = hxp[t + k * TPB];

    auto ln_act = [&](float* p, int gate, bool use_tanh, float* out) {
        float s = 0.f, s2 = 0.f;
#pragma unroll
        for (int k = 0; k < VPT; ++k) { s += p[k]; s2 += p[k] * p[k]; }
        red[t] = s; __syncthreads();
        for (int o = TPB / 2; o > 0; o >>= 1) {
            if (t < o) red[t] += red[t + o];
            __syncthreads();
        }
        const float mean = red[0] * (1.0f / D_); __syncthreads();
        red[t] = s2; __syncthreads();
        for (int o = TPB / 2; o > 0; o >>= 1) {
            if (t < o) red[t] += red[t + o];
            __syncthreads();
        }
        const float var = red[0] * (1.0f / D_) - mean * mean; __syncthreads();
        const float inv = rsqrtf(var + 1e-5f);
#pragma unroll
        for (int k = 0; k < VPT; ++k) {
            const int d = t + k * TPB;
            const float y = (p[k] - mean) * inv * ln_g[gate * D_ + d] + ln_b[gate * D_ + d];
            out[k] = use_tanh ? tanhf(y) : (1.0f / (1.0f + expf(-y)));
        }
    };

#pragma unroll
    for (int k = 0; k < VPT; ++k) { const int d = t + k * TPB; pre[k] = gx[d] + ghb[d]; }
    ln_act(pre, 0, false, r);
#pragma unroll
    for (int k = 0; k < VPT; ++k) { const int d = t + k * TPB; pre[k] = gx[D_ + d] + ghb[D_ + d]; }
    ln_act(pre, 1, false, z);
#pragma unroll
    for (int k = 0; k < VPT; ++k) { const int d = t + k * TPB; pre[k] = gx[2 * D_ + d] + r[k] * ghb[2 * D_ + d]; }
    ln_act(pre, 2, true, n);

    float*  of = gru_f32 + ((size_t)b * S_ + step) * (size_t)D_;
    __bf16* ob = gru_bf  + ((size_t)b * S_ + step) * (size_t)D_;
#pragma unroll
    for (int k = 0; k < VPT; ++k) {
        const int d = t + k * TPB;
        const float h = (1.0f - z[k]) * n[k] + z[k] * hx[k];
        of[d] = h;
        ob[d] = (__bf16)h;
    }
}

// ---------------------------------------------------------------------------
// emb scan: one block per batch b, sequential over S inside the block.
//   emb[b,i] = concat(shorted[b,i] (64), emb[b, fa[b,i], 0:960])
// fa[b,i] < i for i >= 1; i == 0 uses zeros in place of the self-reference.
// ---------------------------------------------------------------------------
__global__ __launch_bounds__(256) void emb_scan_kernel(
    const float* __restrict__ shorted,  // (B,S,DS)
    const int*   __restrict__ fa,
    float*  __restrict__ emb,           // (B,S,D)
    __bf16* __restrict__ emb_bf) {

    constexpr int TPB = 256, VPT = D_ / TPB;
    const int b = blockIdx.x, t = threadIdx.x;
    for (int i = 0; i < S_; ++i) {
        const int parent = fa[b * S_ + i];
        float v[VPT];
#pragma unroll
        for (int k = 0; k < VPT; ++k) {
            const int d = t + k * TPB;
            if (d < DS_)
                v[k] = shorted[((size_t)b * S_ + i) * DS_ + d];
            else
                v[k] = (i == 0) ? 0.0f
                                : emb[((size_t)b * S_ + parent) * (size_t)D_ + (d - DS_)];
        }
#pragma unroll
        for (int k = 0; k < VPT; ++k) {
            const int d = t + k * TPB;
            const size_t idx = ((size_t)b * S_ + i) * (size_t)D_ + d;
            emb[idx]    = v[k];
            emb_bf[idx] = (__bf16)v[k];
        }
        __syncthreads();   // make row i visible before later iterations read it
    }
}

// ---------------------------------------------------------------------------
// Host side
// ---------------------------------------------------------------------------
static inline void gemm(hipStream_t s,
                        const __bf16* A, int lda, const __bf16* Bw, int ldb,
                        const float* bias, float* Cf, __bf16* Cb, int ldc,
                        int M, int N, int K, int accumulate, int act,
                        const int* fa = nullptr, int step = 0, int Sdim = 0) {
    dim3 grid((M + 127) / 128, (N + 127) / 128);
    wmma_gemm_kernel<<<grid, 256, 0, s>>>(A, lda, Bw, ldb, bias, Cf, Cb, ldc,
                                          M, N, K, accumulate, act, fa, step, Sdim);
}

static inline void cvt(hipStream_t s, const float* in, __bf16* out, size_t n) {
    int blocks = (int)(((n + 255) / 256 < 4096) ? (n + 255) / 256 : 4096);
    f32_to_bf16_kernel<<<blocks, 256, 0, s>>>(in, out, n);
}

extern "C" void kernel_launch(void* const* d_in, const int* in_sizes, int n_in,
                              void* d_out, int out_size, void* d_ws, size_t ws_size,
                              hipStream_t stream) {
    (void)in_sizes; (void)n_in; (void)out_size; (void)ws_size;

    const float* token = (const float*)d_in[0];
    const float* Wx    = (const float*)d_in[1];
    const float* bx    = (const float*)d_in[2];
    const float* Wh    = (const float*)d_in[3];
    const float* bh    = (const float*)d_in[4];
    const float* ln_g  = (const float*)d_in[5];
    const float* ln_b  = (const float*)d_in[6];
    const float* fc0_w = (const float*)d_in[7];
    const float* fc0_b = (const float*)d_in[8];
    const float* fc1_w = (const float*)d_in[9];
    const float* fc1_b = (const float*)d_in[10];
    const float* cmb_w = (const float*)d_in[11];
    const float* cmb_b = (const float*)d_in[12];
    const int*   fa    = (const int*)  d_in[13];
    float* out = (float*)d_out;

    const size_t M  = (size_t)B_ * S_;      // 16384
    char* base = (char*)d_ws;
    size_t off = 0;
    auto alloc = [&](size_t bytes) -> char* {
        char* p = base + off;
        off += (bytes + 255) & ~(size_t)255;
        return p;
    };

    // Persistent segments.
    __bf16* tok_bf  = (__bf16*)alloc(M * D_ * 2);                 // 32 MB
    __bf16* Wx_bf   = (__bf16*)alloc((size_t)3 * D_ * D_ * 2);    // 6 MB
    __bf16* Wh_bf   = (__bf16*)alloc((size_t)3 * D_ * D_ * 2);    // 6 MB
    __bf16* fc0_bf  = (__bf16*)alloc((size_t)D_ * D_ * 2);        // 2 MB
    __bf16* fc1_bf  = (__bf16*)alloc((size_t)DS_ * D_ * 2);       // 128 KB
    __bf16* cmb_bf  = (__bf16*)alloc((size_t)D_ * 2 * D_ * 2);    // 4 MB
    float*  gru_f32 = (float*) alloc(M * D_ * 4);                 // 64 MB
    __bf16* gru_bf  = (__bf16*)alloc(M * D_ * 2);                 // 32 MB
    float*  gh_buf  = (float*) alloc((size_t)B_ * 3 * D_ * 4);    // 1.5 MB
    // Arena reused across phases: gx_all (scan phase) vs post-scan buffers.
    char* arena = alloc((size_t)M * 3 * D_ * 4);                  // 192 MB
    float*  gx_all  = (float*)arena;                              // (B,S,3D) during scan
    __bf16* h0_bf   = (__bf16*)arena;                             // 32 MB  (after scan)
    float*  shorted = (float*)(arena + ((M * D_ * 2 + 255) & ~(size_t)255));
    float*  emb_f32 = (float*)((char*)shorted + ((M * DS_ * 4 + 255) & ~(size_t)255));
    __bf16* emb_bf  = (__bf16*)((char*)emb_f32 + ((M * D_ * 4 + 255) & ~(size_t)255));

    // ---- 0) precision conversion + state init -----------------------------
    cvt(stream, token, tok_bf, M * D_);
    cvt(stream, Wx,    Wx_bf,  (size_t)3 * D_ * D_);
    cvt(stream, Wh,    Wh_bf,  (size_t)3 * D_ * D_);
    cvt(stream, fc0_w, fc0_bf, (size_t)D_ * D_);
    cvt(stream, fc1_w, fc1_bf, (size_t)DS_ * D_);
    cvt(stream, cmb_w, cmb_bf, (size_t)D_ * 2 * D_);
    hipMemsetAsync(gru_f32, 0, M * D_ * 4, stream);
    hipMemsetAsync(gru_bf,  0, M * D_ * 2, stream);

    // ---- 1) gx_all = token @ Wx^T + bx  (16384 x 3072 x 1024) -------------
    gemm(stream, tok_bf, D_, Wx_bf, D_, bx, gx_all, nullptr, 3 * D_,
         (int)M, 3 * D_, D_, 0, 0);

    // ---- 2) GRU scan: per step a gathered GEMM + gate kernel --------------
    for (int step = 0; step < S_; ++step) {
        // gh = gru_bf[b, fa[b,step]] @ Wh^T + bh   (128 x 3072 x 1024)
        gemm(stream, gru_bf, D_, Wh_bf, D_, bh, gh_buf, nullptr, 3 * D_,
             B_, 3 * D_, D_, 0, 0, fa, step, S_);
        gru_gate_kernel<<<B_, 256, 0, stream>>>(gx_all, gh_buf, ln_g, ln_b,
                                                fa, step, gru_f32, gru_bf);
    }

    // ---- 3) h0 = gelu(gru @ fc0^T + b)  -> bf16 only ----------------------
    gemm(stream, gru_bf, D_, fc0_bf, D_, fc0_b, nullptr, h0_bf, D_,
         (int)M, D_, D_, 0, 1);

    // ---- 4) shorted = h0 @ fc1^T + b   (N = 64) ---------------------------
    gemm(stream, h0_bf, D_, fc1_bf, D_, fc1_b, shorted, nullptr, DS_,
         (int)M, DS_, D_, 0, 0);

    // ---- 5) emb scan (gather/concat chain) --------------------------------
    emb_scan_kernel<<<B_, 256, 0, stream>>>(shorted, fa, emb_f32, emb_bf);

    // ---- 6) out = token @ cmb[:, :D]^T + emb @ cmb[:, D:]^T + cmb_b -------
    gemm(stream, tok_bf, D_, cmb_bf,      2 * D_, cmb_b,   out, nullptr, D_,
         (int)M, D_, D_, 0, 0);
    gemm(stream, emb_bf, D_, cmb_bf + D_, 2 * D_, nullptr, out, nullptr, D_,
         (int)M, D_, D_, 1, 0);
}
